// EncoderLayer_39290360823991
// MI455X (gfx1250) — matrix operand order repaired and checked by reading, not compiled, not run on hardware
//
#include <hip/hip_runtime.h>
#include <math.h>

// ---------------------------------------------------------------------------
// CDNA5 / gfx1250 encoder layer: BF16 WMMA GEMMs + flash attention.
// B=2, S=2048, D=768, H=12, Dh=64, FF=3072. f32 in/out, bf16 WMMA compute.
// LDS layouts are swizzled so every WMMA fragment loads as 2x ds_load_b128.
// ---------------------------------------------------------------------------

#define D_MODEL 768
#define D_FF    3072
#define SEQ     2048
#define NBATCH  2
#define NHEADS  12
#define NROWS   (NBATCH * SEQ)          // 4096

typedef __attribute__((ext_vector_type(16))) __bf16 bf16x16;
typedef __attribute__((ext_vector_type(8)))  float  v8f;

// D = A(16x32 bf16) * B(32x16 bf16) + C(16x16 f32)
__device__ __forceinline__ v8f wmma_bf16(bf16x16 a, bf16x16 b, v8f c) {
    return __builtin_amdgcn_wmma_f32_16x16x32_bf16(
        false, a, false, b, (short)0, c, false, false);
}

__device__ __forceinline__ void zero_v8f(v8f& v) {
#pragma unroll
    for (int i = 0; i < 8; ++i) v[i] = 0.0f;
}

// Load 16 contiguous bf16 (32B) from LDS as one fragment (2x ds_load_b128).
__device__ __forceinline__ bf16x16 load_frag16(const __bf16* p) {
    union { bf16x16 v; uint4 u[2]; } t;
    t.u[0] = *(const uint4*)p;
    t.u[1] = *(const uint4*)(p + 8);
    return t.v;
}

// Pack 8 f32 -> 8 bf16 and store 16B to LDS (v_cvt_pk_bf16_f32 + ds_store_b128).
__device__ __forceinline__ void store8_bf16(__bf16* dst, const float* s) {
    union { uint4 u; __bf16 h[8]; } t;
#pragma unroll
    for (int i = 0; i < 8; ++i) t.h[i] = (__bf16)s[i];
    *(uint4*)dst = t.u;
}
__device__ __forceinline__ void store8_bf16v(__bf16* dst, const float* s, float scale) {
    union { uint4 u; __bf16 h[8]; } t;
#pragma unroll
    for (int i = 0; i < 8; ++i) t.h[i] = (__bf16)(s[i] * scale);
    *(uint4*)dst = t.u;
}

// A-operand K order (ISA 7.12.2, 16-bit A 16x32): AMD's table has lane-half h hold
// orig K {h*8..h*8+7, 16+h*8..16+h*8+7}. This code instead uses K = 16h+s
// for A and B, so columns stay unpermuted and each half's 16 values are
// contiguous. Store orig cols [c0, c0+16) (c0 % 16 == 0) into natural (contiguous) slots:
__device__ __forceinline__ void store_a_perm16(__bf16* rowbase, int c0, const float* v) {
    const int blk = c0 & ~31;
    const int lo  = (c0 & 16) ? 16 : 0;   // orig c0..c0+7
    const int hi  = (c0 & 16) ? 24 : 8;   // orig c0+8..c0+15
    store8_bf16(rowbase + blk + lo, v);
    store8_bf16(rowbase + blk + hi, v + 8);
}

// ---------------------------------------------------------------------------
// GEMM: C[M,N] = op(A[M,K] @ W[K,N] (+bias) (+relu) (+residual)).
// 256 threads = 8 wave32; block tile 128x128; per-wave 32x64 = 2x4 WMMA tiles.
// Per k-step per wave: 2 A-frags + 4 B-frags (12x ds_load_b128) -> 8 WMMAs.
// M%128==0, N%128==0, K%32==0 (holds for all uses here).
// ---------------------------------------------------------------------------
__global__ __launch_bounds__(256) void gemm_bf16_wmma(
    const float* __restrict__ A, const float* __restrict__ W,
    const float* __restrict__ bias, const float* __restrict__ residual,
    float* __restrict__ C, int M, int N, int K, int relu) {

    __shared__ __align__(16) __bf16 As[128][40];  // A tile [m][k], natural K order, stride 80B
    __shared__ __align__(16) __bf16 Bt[128][40];  // B tile transposed [n][k], stride 80B

    const int tid  = threadIdx.x;
    const int lane = tid & 31;
    const int w    = tid >> 5;       // 0..7
    const int wm   = w >> 1;         // 0..3 : 32-row band
    const int wn   = w & 1;          // 0..1 : 64-col band
    const int hh   = lane >> 4;
    const int l15  = lane & 15;

    const int blockM = blockIdx.y * 128;
    const int blockN = blockIdx.x * 128;

    v8f acc[2][4];
#pragma unroll
    for (int i = 0; i < 2; ++i)
#pragma unroll
        for (int j = 0; j < 4; ++j) zero_v8f(acc[i][j]);

    // Staging thread roles
    const int ar  = tid >> 1;              // A: row 0..127
    const int ac0 = (tid & 1) << 4;        // A: col 0 or 16
    const int bn0 = (tid & 31) << 2;       // B: 4 cols n0..n0+3
    const int bk0 = (tid >> 5) << 2;       // B: 4 rows k0..k0+3

    for (int kk = 0; kk < K; kk += 32) {
        __syncthreads();
        {   // A tile: 16 f32 per thread, contiguous store (2x b128)
            const float* ap = A + (size_t)(blockM + ar) * K + kk + ac0;
            float v[16];
#pragma unroll
            for (int c = 0; c < 16; ++c) v[c] = ap[c];
            store_a_perm16(&As[ar][0], ac0, v);
            if (kk + 32 < K) __builtin_prefetch(ap + 32, 0, 1);  // global_prefetch_b8
        }
        {   // B tile: 4x4 micro-transpose per thread -> Bt[n][k]
            float v[4][4];
#pragma unroll
            for (int i = 0; i < 4; ++i) {
                const float* bp = W + (size_t)(kk + bk0 + i) * N + blockN + bn0;
#pragma unroll
                for (int j = 0; j < 4; ++j) v[i][j] = bp[j];
            }
#pragma unroll
            for (int j = 0; j < 4; ++j) {
                union { uint2 u; __bf16 h[4]; } t;
#pragma unroll
                for (int i = 0; i < 4; ++i) t.h[i] = (__bf16)v[i][j];
                *(uint2*)&Bt[bn0 + j][bk0] = t.u;
            }
        }
        __syncthreads();

        bf16x16 a0 = load_frag16(&As[wm * 32 + l15][hh * 16]);
        bf16x16 a1 = load_frag16(&As[wm * 32 + 16 + l15][hh * 16]);
        bf16x16 bf[4];
#pragma unroll
        for (int ns = 0; ns < 4; ++ns)
            bf[ns] = load_frag16(&Bt[wn * 64 + ns * 16 + l15][hh * 16]);

#pragma unroll
        for (int ns = 0; ns < 4; ++ns) {
            acc[0][ns] = wmma_bf16(a0, bf[ns], acc[0][ns]);
            acc[1][ns] = wmma_bf16(a1, bf[ns], acc[1][ns]);
        }
    }

    // Epilogue: C layout: reg r -> row = r + 8*h, lane%16 -> col.
#pragma unroll
    for (int ms = 0; ms < 2; ++ms) {
#pragma unroll
        for (int ns = 0; ns < 4; ++ns) {
            const int col = blockN + wn * 64 + ns * 16 + l15;
#pragma unroll
            for (int r = 0; r < 8; ++r) {
                const int row = blockM + wm * 32 + ms * 16 + r + 8 * hh;
                float v = acc[ms][ns][r];
                if (bias)     v += bias[col];
                if (relu)     v  = fmaxf(v, 0.0f);
                if (residual) v += residual[(size_t)row * N + col];
                C[(size_t)row * N + col] = v;
            }
        }
    }
}

// ---------------------------------------------------------------------------
// Flash attention: block = (64-query tile) x (batch*head), 256 threads.
// Q tile LDS-resident (natural K order, pre-scaled 1/8); per 64-key block:
// S = Q@K^T via WMMA (K^T operand reads K-tile rows contiguously), online
// softmax (64 row-owner threads), P stored in natural K order, P@V via WMMA
// (V stored transposed [d][key] so the B-operand is K-contiguous).
// ---------------------------------------------------------------------------
__global__ __launch_bounds__(256) void attn_fwd_wmma(
    const float* __restrict__ Qg, const float* __restrict__ Kg,
    const float* __restrict__ Vg, float* __restrict__ Og) {

    __shared__ __align__(16) __bf16 Qs[64][72];  // [q][d]   natural K order, stride 144B
    __shared__ __align__(16) __bf16 Ks[64][72];  // [key][d] natural (B of Q@K^T)
    __shared__ __align__(16) __bf16 Vt[64][72];  // [d][key] transposed (B of P@V)
    __shared__ __align__(16) __bf16 Pb[64][72];  // [q][key] natural K order
    __shared__ float  Sc[64][68];
    __shared__ float  rowscale[64];

    const int tid  = threadIdx.x;
    const int lane = tid & 31;
    const int w    = tid >> 5;
    const int qt   = w >> 1;          // query sub-tile (16 rows)
    const int pr   = w & 1;           // which pair of col sub-tiles
    const int hh   = lane >> 4;
    const int l15  = lane & 15;

    const int b  = blockIdx.y / NHEADS;
    const int h  = blockIdx.y % NHEADS;
    const int q0 = blockIdx.x * 64;

    // Q tile: 16 f32 per thread, scaled by 1/sqrt(64), contiguous store
    {
        const int r  = tid >> 2;
        const int c0 = (tid & 3) << 4;
        const float* p = Qg + ((size_t)(b * SEQ + q0 + r)) * D_MODEL + h * 64 + c0;
        float v[16];
#pragma unroll
        for (int c = 0; c < 16; ++c) v[c] = p[c] * 0.125f;
        store_a_perm16(&Qs[r][0], c0, v);
    }

    float m_run = -INFINITY;   // valid for threads 0..63
    float l_run = 0.0f;
    v8f oacc[2];
    zero_v8f(oacc[0]);
    zero_v8f(oacc[1]);

    for (int kb = 0; kb < SEQ / 64; ++kb) {
        __syncthreads();
        {   // K tile natural [key][d]: 16 f32/thread, contiguous store
            const int r  = tid >> 2;
            const int c0 = (tid & 3) << 4;
            const float* pk = Kg + ((size_t)(b * SEQ + kb * 64 + r)) * D_MODEL + h * 64 + c0;
            float v[16];
#pragma unroll
            for (int c = 0; c < 16; ++c) v[c] = pk[c];
            store8_bf16(&Ks[r][c0], v);
            store8_bf16(&Ks[r][c0 + 8], v + 8);
        }
        {   // V tile transposed [d][key]: 4x4 micro-transpose per thread
            const int key0 = (tid & 15) << 2;
            const int d0   = (tid >> 4) << 2;
            float v[4][4];
#pragma unroll
            for (int i = 0; i < 4; ++i) {
                const float* pv = Vg + ((size_t)(b * SEQ + kb * 64 + key0 + i)) * D_MODEL + h * 64 + d0;
#pragma unroll
                for (int j = 0; j < 4; ++j) v[i][j] = pv[j];
            }
#pragma unroll
            for (int j = 0; j < 4; ++j) {
                union { uint2 u; __bf16 hx[4]; } t;
#pragma unroll
                for (int i = 0; i < 4; ++i) t.hx[i] = (__bf16)v[i][j];
                *(uint2*)&Vt[d0 + j][key0] = t.u;
            }
        }
        __syncthreads();

        // ---- scores: 2 WMMA tiles per wave
#pragma unroll
        for (int ks = 0; ks < 2; ++ks) {
            const int kt = pr * 2 + ks;
            v8f s;
            zero_v8f(s);
#pragma unroll
            for (int kk = 0; kk < 2; ++kk) {
                bf16x16 a  = load_frag16(&Qs[qt * 16 + l15][kk * 32 + hh * 16]);
                bf16x16 bt = load_frag16(&Ks[kt * 16 + l15][kk * 32 + hh * 16]);
                s = wmma_bf16(a, bt, s);
            }
#pragma unroll
            for (int r = 0; r < 8; ++r)
                Sc[qt * 16 + r + 8 * hh][kt * 16 + l15] = s[r];
        }
        __syncthreads();

        // ---- online softmax: thread t owns row t; P written in natural K order
        if (tid < 64) {
            float bm = -INFINITY;
#pragma unroll 8
            for (int c = 0; c < 64; ++c) bm = fmaxf(bm, Sc[tid][c]);
            const float mnew = fmaxf(m_run, bm);
            const float corr = __expf(m_run - mnew);
            float sum = 0.0f;
#pragma unroll
            for (int rb = 0; rb < 8; ++rb) {        // 8-col runs
                float p[8];
#pragma unroll
                for (int c = 0; c < 8; ++c) {
                    p[c] = __expf(Sc[tid][rb * 8 + c] - mnew);
                    sum += p[c];
                }
                static const int map8[4] = {0, 8, 16, 24};
                const int pos = (rb & 4) * 8 + map8[rb & 3];  // (rb/4)*32 + (rb&3)*8 = rb*8
                store8_bf16(&Pb[tid][pos], p);
            }
            l_run = l_run * corr + sum;
            m_run = mnew;
            rowscale[tid] = corr;
        }
        __syncthreads();

        // ---- rescale O, accumulate P @ V
#pragma unroll
        for (int ds = 0; ds < 2; ++ds) {
            const int dt = pr * 2 + ds;
#pragma unroll
            for (int r = 0; r < 8; ++r)
                oacc[ds][r] *= rowscale[qt * 16 + r + 8 * hh];
#pragma unroll
            for (int kk = 0; kk < 2; ++kk) {
                bf16x16 a  = load_frag16(&Pb[qt * 16 + l15][kk * 32 + hh * 16]);
                bf16x16 bv = load_frag16(&Vt[dt * 16 + l15][kk * 32 + hh * 16]);
                oacc[ds] = wmma_bf16(a, bv, oacc[ds]);
            }
        }
    }

    __syncthreads();
    if (tid < 64) rowscale[tid] = 1.0f / l_run;
    __syncthreads();

#pragma unroll
    for (int ds = 0; ds < 2; ++ds) {
        const int dt = pr * 2 + ds;
#pragma unroll
        for (int r = 0; r < 8; ++r) {
            const int row = qt * 16 + r + 8 * hh;
            Og[((size_t)(b * SEQ + q0 + row)) * D_MODEL + h * 64 + dt * 16 + l15] =
                oacc[ds][r] * rowscale[row];
        }
    }
}

// ---------------------------------------------------------------------------
// LayerNorm over last dim (768). One block per row.
// ---------------------------------------------------------------------------
__global__ __launch_bounds__(256) void layernorm_rows(
    const float* __restrict__ x, const float* __restrict__ gamma,
    const float* __restrict__ beta, float* __restrict__ out) {

    __shared__ float rs[256];
    __shared__ float rq[256];

    const int row = blockIdx.x;
    const int tid = threadIdx.x;
    const float* xr = x + (size_t)row * D_MODEL;

    float s = 0.0f, q = 0.0f;
#pragma unroll
    for (int i = 0; i < 3; ++i) {
        const float v = xr[tid + i * 256];
        s += v;
        q += v * v;
    }
    rs[tid] = s;
    rq[tid] = q;
    __syncthreads();
    for (int off = 128; off > 0; off >>= 1) {
        if (tid < off) {
            rs[tid] += rs[tid + off];
            rq[tid] += rq[tid + off];
        }
        __syncthreads();
    }
    const float mean = rs[0] * (1.0f / D_MODEL);
    const float var  = rq[0] * (1.0f / D_MODEL) - mean * mean;
    const float rstd = rsqrtf(var + 1e-5f);

    float* orow = out + (size_t)row * D_MODEL;
#pragma unroll
    for (int i = 0; i < 3; ++i) {
        const int c = tid + i * 256;
        orow[c] = gamma[c] * ((xr[c] - mean) * rstd) + beta[c];
    }
}

// ---------------------------------------------------------------------------
// Host-side orchestration.
// ---------------------------------------------------------------------------
extern "C" void kernel_launch(void* const* d_in, const int* in_sizes, int n_in,
                              void* d_out, int out_size, void* d_ws, size_t ws_size,
                              hipStream_t stream) {
    (void)in_sizes; (void)n_in; (void)out_size; (void)ws_size;

    const float* src    = (const float*)d_in[0];
    const float* Wq     = (const float*)d_in[1];
    const float* Wk     = (const float*)d_in[2];
    const float* Wv     = (const float*)d_in[3];
    const float* Wo     = (const float*)d_in[4];
    const float* W1     = (const float*)d_in[5];
    const float* b1     = (const float*)d_in[6];
    const float* W2     = (const float*)d_in[7];
    const float* b2     = (const float*)d_in[8];
    const float* gamma1 = (const float*)d_in[9];
    const float* beta1  = (const float*)d_in[10];
    const float* gamma2 = (const float*)d_in[11];
    const float* beta2  = (const float*)d_in[12];

    float* ws = (float*)d_ws;
    const size_t T = (size_t)NROWS * D_MODEL;   // 3,145,728 floats
    float* Qb     = ws + 0 * T;
    float* Kb     = ws + 1 * T;
    float* Vb     = ws + 2 * T;
    float* attn   = ws + 3 * T;
    float* hidden = ws + 0 * T;   // 4*T floats; aliases Q/K/V/attn (dead by then)
    float* pre_ln = ws + 4 * T;
    float* x1     = ws + 5 * T;
    // total: 6*T floats = 75.5 MB

    const dim3 blk(256);
    const dim3 gD(D_MODEL / 128, NROWS / 128);  // (6, 32)
    const dim3 gF(D_FF / 128,    NROWS / 128);  // (24, 32)

    // QKV projections
    gemm_bf16_wmma<<<gD, blk, 0, stream>>>(src, Wq, nullptr, nullptr, Qb,
                                           NROWS, D_MODEL, D_MODEL, 0);
    gemm_bf16_wmma<<<gD, blk, 0, stream>>>(src, Wk, nullptr, nullptr, Kb,
                                           NROWS, D_MODEL, D_MODEL, 0);
    gemm_bf16_wmma<<<gD, blk, 0, stream>>>(src, Wv, nullptr, nullptr, Vb,
                                           NROWS, D_MODEL, D_MODEL, 0);

    // Attention
    attn_fwd_wmma<<<dim3(SEQ / 64, NBATCH * NHEADS), blk, 0, stream>>>(Qb, Kb, Vb, attn);

    // Output projection + residual, then LN1
    gemm_bf16_wmma<<<gD, blk, 0, stream>>>(attn, Wo, nullptr, src, pre_ln,
                                           NROWS, D_MODEL, D_MODEL, 0);
    layernorm_rows<<<dim3(NROWS), blk, 0, stream>>>(pre_ln, gamma1, beta1, x1);

    // FFN
    gemm_bf16_wmma<<<gF, blk, 0, stream>>>(x1, W1, b1, nullptr, hidden,
                                           NROWS, D_FF, D_MODEL, 1);
    gemm_bf16_wmma<<<gD, blk, 0, stream>>>(hidden, W2, b2, x1, pre_ln,
                                           NROWS, D_MODEL, D_FF, 0);
    layernorm_rows<<<dim3(NROWS), blk, 0, stream>>>(pre_ln, gamma2, beta2, (float*)d_out);
}
